// ShiftAttentionLayer_75608604278947
// MI455X (gfx1250) — matrix operand order repaired and checked by reading, not compiled, not run on hardware
//
#include <hip/hip_runtime.h>
#include <hip/hip_bf16.h>

#define BB 32
#define TT 8
#define NN 256
#define DD 128
#define BT_ (BB * TT)

typedef __attribute__((ext_vector_type(16))) _Float16 v16h;
typedef __attribute__((ext_vector_type(8)))  _Float16 v8h;
typedef __attribute__((ext_vector_type(8)))  float    v8f;
typedef __attribute__((ext_vector_type(4)))  unsigned int v4u;
typedef __attribute__((ext_vector_type(8)))  int      v8i;
typedef __attribute__((ext_vector_type(4)))  int      v4i;

__device__ __forceinline__ v8f wmma_f16(v16h a, v16h b, v8f c) {
  // D = A(16x32 f16) * B(32x16 f16) + C(16x16 f32)
  return __builtin_amdgcn_wmma_f32_16x16x32_f16(
      /*neg_a=*/false, a, /*neg_b=*/false, b,
      /*c_mod=*/(short)0, c, /*reuse_a=*/false, /*reuse_b=*/false);
}

// A fragment: row m (= lane&15), K-block [k0,k0+32), khf = lane>>4.
// Element j holds K = khf*16 + j.
__device__ __forceinline__ v16h load_a16(const _Float16* A, int ld, int m, int k0, int khf) {
  return *(const v16h*)(A + (size_t)m * ld + k0 + khf * 16);
}

// B fragment where B[k][n] = M[n][k] (M row-major): one contiguous 32B load.
// lane col n = lane&15, half-wave khf gives K group of 16: K = khf*16 + j.
__device__ __forceinline__ v16h load_bT16(const _Float16* M, int ld, int n, int k0, int khf) {
  return *(const v16h*)(M + (size_t)n * ld + k0 + khf * 16);
}

__device__ __forceinline__ float wsum(float x) {
#pragma unroll
  for (int m = 16; m >= 1; m >>= 1) x += __shfl_xor(x, m, 32);
  return x;
}
__device__ __forceinline__ float wmaxr(float x) {
#pragma unroll
  for (int m = 16; m >= 1; m >>= 1) x = fmaxf(x, __shfl_xor(x, m, 32));
  return x;
}

// ---- Tensor Data Mover: 2-D f16 tile (dim0 cols x dim1 rows, packed) -> LDS ----
// Descriptor per CDNA5 ISA ch.8: group0 = {count/flags, lds_addr, global_addr, type=2},
// group1 = {data_size=2B, tensor_dim0/1, tile_dim0/1, stride0}.
__device__ __forceinline__ void tdm_load_2d_f16(const void* gsrc, unsigned lds_off,
                                                int dim0, int dim1) {
  unsigned long long ga = (unsigned long long)(size_t)gsrc;
  v4u g0;
  g0[0] = 1u;                       // count=1, user descriptor
  g0[1] = lds_off;                  // lds_addr (bytes)
  g0[2] = (unsigned)ga;             // global_addr[31:0]
  g0[3] = (unsigned)((ga >> 32) & 0x1FFFFFFu) | (2u << 30);  // addr[56:32] | type=2
  v8i g1;
  g1[0] = 0x00010000;               // data_size = 1 (2 bytes); no pad/iterate/multicast
  g1[1] = (dim0 & 0xFFFF) << 16;                      // tensor_dim0[15:0]
  g1[2] = ((dim0 >> 16) & 0xFFFF) | ((dim1 & 0xFFFF) << 16);  // dim0 hi | tensor_dim1 lo
  g1[3] = ((dim1 >> 16) & 0xFFFF) | ((dim0 & 0xFFFF) << 16);  // dim1 hi | tile_dim0
  g1[4] = dim1 & 0xFFFF;            // tile_dim1 | tile_dim2=0
  g1[5] = dim0;                     // tensor_dim0_stride[31:0]
  g1[6] = 0;
  g1[7] = 0;
  v4i gz = {};
#if defined(__clang_major__) && __clang_major__ >= 23
  v8i gz8 = {};
  __builtin_amdgcn_tensor_load_to_lds(g0, g1, gz, gz, gz8, 0);
#else
  __builtin_amdgcn_tensor_load_to_lds(g0, g1, gz, gz, 0);
#endif
}

// ---------------- Kernel 0: f16-transposed weight prep ----------------
__global__ __launch_bounds__(256) void prep_kernel(
    const float* __restrict__ Wq, const float* __restrict__ Wk,
    const float* __restrict__ Wv, const float* __restrict__ W1,
    const float* __restrict__ W2,
    _Float16* __restrict__ WqT, _Float16* __restrict__ WkT,
    _Float16* __restrict__ WvT, _Float16* __restrict__ W1T,
    _Float16* __restrict__ W2T) {
  int sel = blockIdx.y;
  int i = blockIdx.x * 256 + threadIdx.x;
  if (sel < 3) {
    if (i >= TT * DD * DD) return;
    int t = i / (DD * DD);
    int r = i % (DD * DD);
    int e = r / DD, d = r % DD;  // dst[t][e][d] = src[t][d][e]
    const float* S = sel == 0 ? Wq : (sel == 1 ? Wk : Wv);
    _Float16* Dst = sel == 0 ? WqT : (sel == 1 ? WkT : WvT);
    Dst[i] = (_Float16)S[(size_t)t * DD * DD + (size_t)d * DD + e];
  } else if (sel == 3) {
    if (i >= 2 * DD * DD) return;        // W1T[j][k] = W1[k][j], j<256, k<128
    int j = i / DD, k = i % DD;
    W1T[i] = (_Float16)W1[(size_t)k * (2 * DD) + j];
  } else {
    if (i >= 2 * DD * DD) return;        // W2T[j][k] = W2[k][j], j<128, k<256
    int j = i / (2 * DD), k = i % (2 * DD);
    W2T[i] = (_Float16)W2[(size_t)k * DD + j];
  }
}

// ---------------- Kernel 1: q/k/v projections (vT stored transposed) ----------------
__global__ __launch_bounds__(256) void qkv_kernel(
    const float* __restrict__ x, const float* __restrict__ tf,
    const _Float16* __restrict__ WqT, const float* __restrict__ bq,
    const _Float16* __restrict__ WkT, const float* __restrict__ bk,
    const _Float16* __restrict__ WvT, const float* __restrict__ bv,
    _Float16* __restrict__ qh, _Float16* __restrict__ khb, _Float16* __restrict__ vT) {
  __shared__ __align__(32) _Float16 sxt[16 * DD];
  int bt = blockIdx.x;
  int t = bt % TT;
  int m0 = blockIdx.y * 16;
  int tid = threadIdx.x;
  const float* xr = x + (size_t)bt * NN * DD;
  const float* tr = tf + (size_t)bt * DD;
  for (int i = tid; i < 16 * DD; i += 256) {
    int r = i >> 7, c = i & (DD - 1);
    sxt[i] = (_Float16)(xr[(size_t)(m0 + r) * DD + c] + tr[c]);
  }
  __syncthreads();
  int lane = tid & 31, wave = tid >> 5;
  int col = lane & 15, khf = lane >> 4;
  int n0 = wave * 16;  // 8 waves x 16 cols = 128 output cols
  const _Float16* wq = WqT + (size_t)t * DD * DD;
  const _Float16* wk = WkT + (size_t)t * DD * DD;
  const _Float16* wv = WvT + (size_t)t * DD * DD;
  v8f aq = {}, ak = {}, av = {};
#pragma unroll
  for (int ks = 0; ks < 4; ++ks) {
    int k0 = ks * 32;
    v16h a = load_a16(sxt, DD, col, k0, khf);
    v16h bqf = load_bT16(wq, DD, n0 + col, k0, khf);
    v16h bkf = load_bT16(wk, DD, n0 + col, k0, khf);
    v16h bvf = load_bT16(wv, DD, n0 + col, k0, khf);
    aq = wmma_f16(a, bqf, aq);
    ak = wmma_f16(a, bkf, ak);
    av = wmma_f16(a, bvf, av);
  }
  float biq = bq[t * DD + n0 + col];
  float bik = bk[t * DD + n0 + col];
  float biv = bv[t * DD + n0 + col];
#pragma unroll
  for (int r = 0; r < 8; ++r) {
    int row = m0 + r + 8 * khf;
    size_t o = ((size_t)bt * NN + row) * DD + n0 + col;
    qh[o] = (_Float16)(aq[r] + biq);
    khb[o] = (_Float16)(ak[r] + bik);
    // v stored transposed: vT[bt][d][m]
    vT[((size_t)bt * DD + n0 + col) * NN + row] = (_Float16)(av[r] + biv);
  }
}

// ---------------- Kernel 2: w = softmax(q k^T / sqrt(D)); emits w, wT, uw ----------------
__global__ __launch_bounds__(256) void attn_w_kernel(
    const _Float16* __restrict__ qh, const _Float16* __restrict__ khb,
    _Float16* __restrict__ wb, _Float16* __restrict__ wT,
    _Float16* __restrict__ uw) {
  __shared__ float sf[16 * NN];
  int bt = blockIdx.x;
  int m0 = blockIdx.y * 16;
  int tid = threadIdx.x, lane = tid & 31, wave = tid >> 5;
  int col = lane & 15, khf = lane >> 4;
  int n0 = wave * 32;  // 8 waves x 32 cols = 256
  const _Float16* Q = qh + (size_t)bt * NN * DD;
  const _Float16* K = khb + (size_t)bt * NN * DD;
  v8f c0 = {}, c1 = {};
#pragma unroll
  for (int ks = 0; ks < 4; ++ks) {
    int k0 = ks * 32;
    v16h a = load_a16(Q, DD, m0 + col, k0, khf);
    v16h b0 = load_bT16(K, DD, n0 + col, k0, khf);
    v16h b1 = load_bT16(K, DD, n0 + 16 + col, k0, khf);
    c0 = wmma_f16(a, b0, c0);
    c1 = wmma_f16(a, b1, c1);
  }
  const float sc = 0.08838834764831845f;  // 1/sqrt(128)
#pragma unroll
  for (int r = 0; r < 8; ++r) {
    int row = r + 8 * khf;
    sf[row * NN + n0 + col] = c0[r] * sc;
    sf[row * NN + n0 + 16 + col] = c1[r] * sc;
  }
  __syncthreads();
#pragma unroll
  for (int rr = 0; rr < 2; ++rr) {
    int row = wave * 2 + rr;
    const float* sr = sf + row * NN;
    float vals[8];
    float vmax = -3.0e38f;
#pragma unroll
    for (int i = 0; i < 8; ++i) { vals[i] = sr[lane + 32 * i]; vmax = fmaxf(vmax, vals[i]); }
    vmax = wmaxr(vmax);
    float sum = 0.f;
#pragma unroll
    for (int i = 0; i < 8; ++i) { vals[i] = __expf(vals[i] - vmax); sum += vals[i]; }
    sum = wsum(sum);
    float inv = 1.0f / sum;
    int grow = m0 + row;
    _Float16* wo = wb + ((size_t)bt * NN + grow) * NN;
    _Float16* uo = uw + ((size_t)bt * NN + grow) * NN;
    _Float16* wt = wT + (size_t)bt * NN * NN;
#pragma unroll
    for (int i = 0; i < 8; ++i) {
      int c = lane + 32 * i;
      _Float16 h = (_Float16)(vals[i] * inv);
      wo[c] = h;
      uo[c] = h;
      wt[(size_t)c * NN + grow] = h;
    }
  }
}

// ------- Kernel 3 (fused): sw = softmax(w_prev w_cur^T/sqrt(N)); uw = sw w_cur + w_cur -------
__global__ __launch_bounds__(256) void shift_kernel(
    const _Float16* __restrict__ wb, const _Float16* __restrict__ wT,
    _Float16* __restrict__ uw) {
  __shared__ float sf[16 * NN];
  __shared__ __align__(32) _Float16 sA[16 * NN];
  int idx = blockIdx.x;
  int b = idx / (TT - 1);
  int t = idx % (TT - 1) + 1;
  int btc = b * TT + t;
  const _Float16* Wp = wb + (size_t)(btc - 1) * NN * NN;   // w_prev (row major)
  const _Float16* Wc = wb + (size_t)btc * NN * NN;         // w_cur  (row major)
  const _Float16* WcT = wT + (size_t)btc * NN * NN;        // w_cur transposed
  int m0 = blockIdx.y * 16;
  int tid = threadIdx.x, lane = tid & 31, wave = tid >> 5;
  int col = lane & 15, khf = lane >> 4;
  int n0 = wave * 32;
  // phase 1: sw scores (reduction over m; B = w_cur^T -> rows of Wc)
  v8f c0 = {}, c1 = {};
#pragma unroll
  for (int ks = 0; ks < 8; ++ks) {
    int k0 = ks * 32;
    v16h a = load_a16(Wp, NN, m0 + col, k0, khf);
    v16h b0 = load_bT16(Wc, NN, n0 + col, k0, khf);
    v16h b1 = load_bT16(Wc, NN, n0 + 16 + col, k0, khf);
    c0 = wmma_f16(a, b0, c0);
    c1 = wmma_f16(a, b1, c1);
  }
  const float scn = 0.0625f;  // 1/sqrt(256)
#pragma unroll
  for (int r = 0; r < 8; ++r) {
    int row = r + 8 * khf;
    sf[row * NN + n0 + col] = c0[r] * scn;
    sf[row * NN + n0 + 16 + col] = c1[r] * scn;
  }
  __syncthreads();
#pragma unroll
  for (int rr = 0; rr < 2; ++rr) {
    int row = wave * 2 + rr;
    const float* sr = sf + row * NN;
    float vals[8];
    float vmax = -3.0e38f;
#pragma unroll
    for (int i = 0; i < 8; ++i) { vals[i] = sr[lane + 32 * i]; vmax = fmaxf(vmax, vals[i]); }
    vmax = wmaxr(vmax);
    float sum = 0.f;
#pragma unroll
    for (int i = 0; i < 8; ++i) { vals[i] = __expf(vals[i] - vmax); sum += vals[i]; }
    sum = wsum(sum);
    float inv = 1.0f / sum;
#pragma unroll
    for (int i = 0; i < 8; ++i) sA[row * NN + lane + 32 * i] = (_Float16)(vals[i] * inv);
  }
  __syncthreads();
  // phase 2: new_w = sA(LDS) @ w_cur + w_cur ; B[k][n] = Wc[k][n] = WcT[n][k]
  v8f d0 = {}, d1 = {};
#pragma unroll
  for (int ks = 0; ks < 8; ++ks) {
    int k0 = ks * 32;
    v16h a = load_a16(sA, NN, col, k0, khf);
    v16h b0 = load_bT16(WcT, NN, n0 + col, k0, khf);
    v16h b1 = load_bT16(WcT, NN, n0 + 16 + col, k0, khf);
    d0 = wmma_f16(a, b0, d0);
    d1 = wmma_f16(a, b1, d1);
  }
  _Float16* U = uw + (size_t)btc * NN * NN;
#pragma unroll
  for (int r = 0; r < 8; ++r) {
    int row = m0 + r + 8 * khf;
    int ca = n0 + col, cb = n0 + 16 + col;
    U[(size_t)row * NN + ca] = (_Float16)(d0[r] + (float)Wc[(size_t)row * NN + ca]);
    U[(size_t)row * NN + cb] = (_Float16)(d1[r] + (float)Wc[(size_t)row * NN + cb]);
  }
}

// ---------------- Kernel 4: out = uw v; ffn_in = LN(out + x). vT staged via TDM ----------------
__global__ __launch_bounds__(512) void out_ln_kernel(
    const _Float16* __restrict__ uw, const _Float16* __restrict__ vT,
    const float* __restrict__ x, const float* __restrict__ ln_g,
    const float* __restrict__ ln_b, float* __restrict__ f32o,
    _Float16* __restrict__ f16o) {
  __shared__ __align__(32) _Float16 sv[DD * NN];  // vT tile [d][m], 64KB (reused for LN scratch)
  int bt = blockIdx.x;
  int tid = threadIdx.x, lane = tid & 31, wave = tid >> 5;
  int col = lane & 15, khf = lane >> 4;
  int m0 = wave * 16;  // 16 waves cover 256 rows
  if (wave == 0) {
    tdm_load_2d_f16(vT + (size_t)bt * DD * NN, (unsigned)(size_t)(void*)sv, NN, DD);
    __builtin_amdgcn_s_wait_tensorcnt(0);
  }
  __syncthreads();
  const _Float16* A = uw + (size_t)bt * NN * NN;
  v8f zero = {};
  v8f acc[8];
#pragma unroll
  for (int i = 0; i < 8; ++i) acc[i] = zero;
#pragma unroll
  for (int ks = 0; ks < 8; ++ks) {
    int k0 = ks * 32;
    v16h a = load_a16(A, NN, m0 + col, k0, khf);
#pragma unroll
    for (int ct = 0; ct < 8; ++ct) {
      // B[m][d] = v[m][d] = sv[d*NN + m]
      v16h bf = load_bT16(sv, NN, ct * 16 + col, k0, khf);
      acc[ct] = wmma_f16(a, bf, acc[ct]);
    }
  }
  const float* xr = x + (size_t)bt * NN * DD;
  float s = 0.f, s2 = 0.f;
#pragma unroll
  for (int ct = 0; ct < 8; ++ct)
#pragma unroll
    for (int r = 0; r < 8; ++r) {
      int row = m0 + r + 8 * khf, c = ct * 16 + col;
      float z = acc[ct][r] + xr[(size_t)row * DD + c];
      acc[ct][r] = z;
      s += z; s2 += z * z;
    }
  s = wsum(s); s2 = wsum(s2);
  __syncthreads();               // sv reads complete; reuse as reduction scratch
  float* red = (float*)sv;
  if (lane == 0) { red[wave] = s; red[16 + wave] = s2; }
  __syncthreads();
  if (tid == 0) {
    float a0 = 0.f, a1 = 0.f;
    for (int i = 0; i < 16; ++i) { a0 += red[i]; a1 += red[16 + i]; }
    red[32] = a0; red[33] = a1;
  }
  __syncthreads();
  const float invC = 1.0f / (float)(NN * DD);
  float mu = red[32] * invC;
  float var = red[33] * invC - mu * mu;
  float rstd = rsqrtf(var + 1e-5f);
#pragma unroll
  for (int ct = 0; ct < 8; ++ct)
#pragma unroll
    for (int r = 0; r < 8; ++r) {
      int row = m0 + r + 8 * khf, c = ct * 16 + col;
      float nz = (acc[ct][r] - mu) * rstd * ln_g[(size_t)row * DD + c] + ln_b[(size_t)row * DD + c];
      size_t o = ((size_t)bt * NN + row) * DD + c;
      f32o[o] = nz;
      f16o[o] = (_Float16)nz;
    }
}

// ---------------- Kernel 5: h = prelu(ffn_in W1 + b1) ----------------
__global__ __launch_bounds__(256) void ffn1_kernel(
    const _Float16* __restrict__ f16i, const _Float16* __restrict__ W1T,
    const float* __restrict__ b1, const float* __restrict__ pa,
    _Float16* __restrict__ hb) {
  int bt = blockIdx.x, m0 = blockIdx.y * 16;
  int tid = threadIdx.x, lane = tid & 31, wave = tid >> 5;
  int col = lane & 15, khf = lane >> 4;
  int n0 = wave * 32;  // 8 waves x 32 = 256 cols
  const _Float16* A = f16i + (size_t)bt * NN * DD;
  v8f c0 = {}, c1 = {};
#pragma unroll
  for (int ks = 0; ks < 4; ++ks) {
    int k0 = ks * 32;
    v16h a = load_a16(A, DD, m0 + col, k0, khf);
    v16h b0 = load_bT16(W1T, DD, n0 + col, k0, khf);
    v16h b1f = load_bT16(W1T, DD, n0 + 16 + col, k0, khf);
    c0 = wmma_f16(a, b0, c0);
    c1 = wmma_f16(a, b1f, c1);
  }
  float alpha = pa[0];
  float bi0 = b1[n0 + col], bi1 = b1[n0 + 16 + col];
  _Float16* H = hb + (size_t)bt * NN * (2 * DD);
#pragma unroll
  for (int r = 0; r < 8; ++r) {
    int row = m0 + r + 8 * khf;
    float h0 = c0[r] + bi0; h0 = h0 >= 0.f ? h0 : alpha * h0;
    float h1 = c1[r] + bi1; h1 = h1 >= 0.f ? h1 : alpha * h1;
    H[(size_t)row * (2 * DD) + n0 + col] = (_Float16)h0;
    H[(size_t)row * (2 * DD) + n0 + 16 + col] = (_Float16)h1;
  }
}

// ---------------- Kernel 6: out = LN(h W2 + b2 + ffn_in). W2T staged via TDM ----------------
__global__ __launch_bounds__(512) void ffn2_ln_kernel(
    const _Float16* __restrict__ hb, const _Float16* __restrict__ W2T,
    const float* __restrict__ b2, const float* __restrict__ f32i,
    const float* __restrict__ ln_g, const float* __restrict__ ln_b,
    float* __restrict__ out) {
  __shared__ __align__(32) _Float16 sw2[DD * 2 * DD];  // W2T [j][k], 64KB (reused for LN scratch)
  int bt = blockIdx.x;
  int tid = threadIdx.x, lane = tid & 31, wave = tid >> 5;
  int col = lane & 15, khf = lane >> 4;
  int m0 = wave * 16;
  if (wave == 0) {
    tdm_load_2d_f16(W2T, (unsigned)(size_t)(void*)sw2, 2 * DD, DD);
    __builtin_amdgcn_s_wait_tensorcnt(0);
  }
  __syncthreads();
  const _Float16* A = hb + (size_t)bt * NN * (2 * DD);
  v8f zero = {};
  v8f acc[8];
#pragma unroll
  for (int i = 0; i < 8; ++i) acc[i] = zero;
#pragma unroll
  for (int ks = 0; ks < 8; ++ks) {
    int k0 = ks * 32;
    v16h a = load_a16(A, 2 * DD, m0 + col, k0, khf);
#pragma unroll
    for (int ct = 0; ct < 8; ++ct) {
      // B[k][j] = W2[k][j] = sw2[j*256 + k]
      v16h bf = load_bT16(sw2, 2 * DD, ct * 16 + col, k0, khf);
      acc[ct] = wmma_f16(a, bf, acc[ct]);
    }
  }
  const float* F = f32i + (size_t)bt * NN * DD;
  float s = 0.f, s2 = 0.f;
#pragma unroll
  for (int ct = 0; ct < 8; ++ct)
#pragma unroll
    for (int r = 0; r < 8; ++r) {
      int row = m0 + r + 8 * khf, c = ct * 16 + col;
      float z = acc[ct][r] + b2[c] + F[(size_t)row * DD + c];
      acc[ct][r] = z;
      s += z; s2 += z * z;
    }
  s = wsum(s); s2 = wsum(s2);
  __syncthreads();               // sw2 reads complete; reuse as reduction scratch
  float* red = (float*)sw2;
  if (lane == 0) { red[wave] = s; red[16 + wave] = s2; }
  __syncthreads();
  if (tid == 0) {
    float a0 = 0.f, a1 = 0.f;
    for (int i = 0; i < 16; ++i) { a0 += red[i]; a1 += red[16 + i]; }
    red[32] = a0; red[33] = a1;
  }
  __syncthreads();
  const float invC = 1.0f / (float)(NN * DD);
  float mu = red[32] * invC;
  float var = red[33] * invC - mu * mu;
  float rstd = rsqrtf(var + 1e-5f);
#pragma unroll
  for (int ct = 0; ct < 8; ++ct)
#pragma unroll
    for (int r = 0; r < 8; ++r) {
      int row = m0 + r + 8 * khf, c = ct * 16 + col;
      float nz = (acc[ct][r] - mu) * rstd * ln_g[(size_t)row * DD + c] + ln_b[(size_t)row * DD + c];
      out[((size_t)bt * NN + row) * DD + c] = nz;
    }
}

extern "C" void kernel_launch(void* const* d_in, const int* in_sizes, int n_in,
                              void* d_out, int out_size, void* d_ws, size_t ws_size,
                              hipStream_t stream) {
  (void)in_sizes; (void)n_in; (void)out_size; (void)ws_size;
  const float* x    = (const float*)d_in[0];
  const float* tf   = (const float*)d_in[1];
  // d_in[2] = adj_mx (unused by reference)
  const float* Wq   = (const float*)d_in[3];
  const float* bq   = (const float*)d_in[4];
  const float* Wk   = (const float*)d_in[5];
  const float* bk   = (const float*)d_in[6];
  const float* Wv   = (const float*)d_in[7];
  const float* bv   = (const float*)d_in[8];
  const float* ln_g = (const float*)d_in[9];
  const float* ln_b = (const float*)d_in[10];
  const float* W1   = (const float*)d_in[11];
  const float* b1   = (const float*)d_in[12];
  const float* pa   = (const float*)d_in[13];
  const float* W2   = (const float*)d_in[14];
  const float* b2   = (const float*)d_in[15];
  float* out = (float*)d_out;

  char* base = (char*)d_ws;
  size_t off = 0;
  auto take = [&](size_t bytes) -> char* {
    off = (off + 255) & ~(size_t)255;
    char* p = base + off;
    off += bytes;
    return p;
  };
  const size_t ND  = (size_t)BT_ * NN * DD;   // 8.4M elems
  const size_t NNE = (size_t)BT_ * NN * NN;   // 16.8M elems
  _Float16* qh   = (_Float16*)take(ND * 2);
  _Float16* khb  = (_Float16*)take(ND * 2);
  _Float16* vT   = (_Float16*)take(ND * 2);
  _Float16* wb   = (_Float16*)take(NNE * 2);
  _Float16* wT   = (_Float16*)take(NNE * 2);
  _Float16* uw   = (_Float16*)take(NNE * 2);
  float*    f32m = (float*)take(ND * 4);
  _Float16* f16m = (_Float16*)take(ND * 2);
  _Float16* hbuf = (_Float16*)take((size_t)BT_ * NN * (2 * DD) * 2);
  _Float16* WqT  = (_Float16*)take((size_t)TT * DD * DD * 2);
  _Float16* WkT  = (_Float16*)take((size_t)TT * DD * DD * 2);
  _Float16* WvT  = (_Float16*)take((size_t)TT * DD * DD * 2);
  _Float16* W1T  = (_Float16*)take((size_t)2 * DD * DD * 2);
  _Float16* W2T  = (_Float16*)take((size_t)2 * DD * DD * 2);

  prep_kernel<<<dim3(512, 5), 256, 0, stream>>>(Wq, Wk, Wv, W1, W2, WqT, WkT, WvT, W1T, W2T);
  qkv_kernel<<<dim3(BT_, NN / 16), 256, 0, stream>>>(x, tf, WqT, bq, WkT, bk, WvT, bv, qh, khb, vT);
  attn_w_kernel<<<dim3(BT_, NN / 16), 256, 0, stream>>>(qh, khb, wb, wT, uw);
  shift_kernel<<<dim3(BB * (TT - 1), NN / 16), 256, 0, stream>>>(wb, wT, uw);
  out_ln_kernel<<<BT_, 512, 0, stream>>>(uw, vT, x, ln_g, ln_b, f32m, f16m);
  ffn1_kernel<<<dim3(BT_, NN / 16), 256, 0, stream>>>(f16m, W1T, b1, pa, hbuf);
  ffn2_ln_kernel<<<BT_, 512, 0, stream>>>(hbuf, W2T, b2, f32m, ln_g, ln_b, out);
}
